// HINGCN_GS_62526133895577
// MI455X (gfx1250) — matrix operand order repaired and checked by reading, not compiled, not run on hardware
//
#include <hip/hip_runtime.h>
#include <hip/hip_bf16.h>
#include <math.h>

typedef __attribute__((ext_vector_type(16))) __bf16 v16bf;
typedef __attribute__((ext_vector_type(8)))  float  v8f;

#define S_SAMP 16
#define DEG_C  32
#define D_FEAT 128
#define E_FEAT 32

__device__ __forceinline__ unsigned pk2(float a, float b) {
  union { __bf16 h[2]; unsigned u; } x;
  x.h[0] = (__bf16)a; x.h[1] = (__bf16)b;
  return x.u;
}

// dst[i,:] = src[idx[i],:]  in float4 units (row = 1<<c4log2 float4s)
__global__ void gather_rows4_k(const float4* __restrict__ src, const int* __restrict__ idx,
                               float4* __restrict__ dst, int rows, int c4log2) {
  int t = blockIdx.x * blockDim.x + threadIdx.x;
  if (t >= (rows << c4log2)) return;
  int i = t >> c4log2, c = t & ((1 << c4log2) - 1);
  dst[t] = src[((long)idx[i] << c4log2) + c];
}

// nb[i*S+s] = adjN[cur[i]*DEG+s], eid likewise (first S columns of DEG)
__global__ void sample_k(const int* __restrict__ cur, int n,
                         const int* __restrict__ adjN, const int* __restrict__ adjE,
                         int* __restrict__ nb, int* __restrict__ eid) {
  int t = blockIdx.x * blockDim.x + threadIdx.x;
  if (t >= n * S_SAMP) return;
  int i = t >> 4, s = t & 15;
  int node = cur[i];
  nb[t]  = adjN[(long)node * DEG_C + s];
  eid[t] = adjE[(long)node * DEG_C + s];
}

// out[i,0:128] = mean_s fsrc[fr(i,s),:], out[i,128:160] = mean_s esrc[er(i,s),:]
// 64 threads/row: wave0 lanes = 32 float4 feature cols, wave1 lanes 0..7 = 8 float4 edge cols.
// fidx/eidx == nullptr -> identity mapping (row i*S+s)
__global__ void mean_concat_k(const float* __restrict__ fsrc, const int* __restrict__ fidx,
                              const float* __restrict__ esrc, const int* __restrict__ eidx,
                              float* __restrict__ out) {
  int i = blockIdx.x, t = threadIdx.x;     // 64 threads
  int sr = i * S_SAMP;
  float4 a = make_float4(0.f, 0.f, 0.f, 0.f);
  if (t < 32) {
    for (int s = 0; s < S_SAMP; ++s) {
      int fr = fidx ? fidx[sr + s] : sr + s;
      if (fidx && s + 1 < S_SAMP) {        // hide L2 gather latency (global_prefetch_b8)
        int fr2 = fidx[sr + s + 1];
        __builtin_prefetch(fsrc + ((long)fr2 << 7) + (t << 2), 0, 0);
      }
      float4 v = ((const float4*)(fsrc + ((long)fr << 7)))[t];
      a.x += v.x; a.y += v.y; a.z += v.z; a.w += v.w;
    }
    ((float4*)(out + (long)i * 160))[t] =
        make_float4(a.x * (1.f / 16.f), a.y * (1.f / 16.f),
                    a.z * (1.f / 16.f), a.w * (1.f / 16.f));
  } else if (t < 40) {
    int te = t - 32;
    for (int s = 0; s < S_SAMP; ++s) {
      int er = eidx ? eidx[sr + s] : sr + s;
      float4 v = ((const float4*)(esrc + ((long)er << 5)))[te];
      a.x += v.x; a.y += v.y; a.z += v.z; a.w += v.w;
    }
    ((float4*)(out + (long)i * 160 + 128))[te] =
        make_float4(a.x * (1.f / 16.f), a.y * (1.f / 16.f),
                    a.z * (1.f / 16.f), a.w * (1.f / 16.f));
  }
}

// C[M,N] = relu( concat_cols(A0>>sA0, A1>>sA1, A2>>sA2) @ concat_rows(B0, B1) )
// A segment s: row-width KAs, output row r reads source row (r >> sAs)  (rep = 1<<sAs).
// All segment boundaries are multiples of 32, so segment choice is UNIFORM per K-step
// and hoisted to scalar code. N in {32,128}. 256 threads = 8 wave32;
// each wave: one 16-col tile, two 16-row tiles (shared B fragment).
//   N=128: Mtile=32 ;  N=32: Mtile=128.  Grid must satisfy gridDim.x*Mtile == M.
__global__ void gemm_relu_wmma_k(
    const float* __restrict__ A0, int KA0, int sA0,
    const float* __restrict__ A1, int KA1, int sA1,
    const float* __restrict__ A2, int KA2, int sA2,
    const float* __restrict__ B0, int KB0,
    const float* __restrict__ B1,
    float* __restrict__ C, int M, int N) {
  __shared__ __bf16 As[128][36];     // [Mtile][Ktile] (+pad: quad stores 8B-aligned)
  __shared__ __bf16 BsT[128][36];    // [N][Ktile] transposed (+pad)

  const int KT     = KA0 + KA1 + KA2;
  const int nTiles = N >> 4;                  // 8 (N=128) or 2 (N=32)
  const int Mtile  = (8 / nTiles) << 5;       // 32 or 128
  const int mBlock = blockIdx.x * Mtile;
  const int nl     = __builtin_ctz(N);

  const int tid  = threadIdx.x;
  const int wave = tid >> 5;
  const int lane = tid & 31;
  const int hf   = lane >> 4;
  const int lr   = lane & 15;
  const int waveM = wave / nTiles;            // 0 or 0..3
  const int waveN = wave - waveM * nTiles;
  const int m0    = (waveM << 5);
  const int m1    = m0 + 16;
  const int nBase = waveN << 4;

  v8f c0 = {}, c1 = {};

  for (int kt = 0; kt < KT; kt += 32) {
    // ---- uniform (scalar) segment selection for this K-step ----
    const float* pA; int strideA, shA, kOffA;
    if (kt < KA0)            { pA = A0; strideA = KA0; shA = sA0; kOffA = kt; }
    else if (kt < KA0 + KA1) { pA = A1; strideA = KA1; shA = sA1; kOffA = kt - KA0; }
    else                     { pA = A2; strideA = KA2; shA = sA2; kOffA = kt - KA0 - KA1; }
    const float* pB; int kOffB;
    if (kt < KB0) { pB = B0; kOffB = kt; }
    else          { pB = B1; kOffB = kt - KB0; }

    // stage A tile [Mtile x 32]: float4 global load -> packed bf16 -> one uint2 LDS store
    for (int e = tid; e < Mtile * 8; e += 256) {
      int row = e >> 3, kp = (e & 7) << 2;    // k offsets kp..kp+3 (quad-aligned)
      int gRow = mBlock + row;
      float4 v = *(const float4*)(pA + (long)(gRow >> shA) * strideA + kOffA + kp);
      *(uint2*)&As[row][kp] = make_uint2(pk2(v.x, v.y), pk2(v.z, v.w));
    }
    // stage B tile transposed: pack k-pairs, one b32 LDS store per thread-item
    for (int e = tid; e < (N << 4); e += 256) {
      int n = e & (N - 1);
      int kp = (e >> nl) << 1;                // even
      const float* q = pB + (long)(kOffB + kp) * N + n;
      *(unsigned*)&BsT[n][kp] = pk2(q[0], q[N]);
    }
    __syncthreads();

    // fragments: adjacent bf16 pairs fetched as dwords (merge to ds_load_2addr_b64)
    union U { v16bf v; unsigned int u[8]; };
    U a0, a1, bu;
#pragma unroll
    for (int j = 0; j < 8; ++j) {
      // A 16x32 bf16: lanes 0-15 rows M; VGPR 0-3 -> K 0..7 (+8*hf), VGPR 4-7 -> K 16..23 (+8*hf)
      int klA = ((j < 4) ? (2 * j) : (16 + 2 * (j - 4))) + 8 * hf;  // even
      a0.u[j] = *(const unsigned int*)&As[m0 + lr][klA];
      a1.u[j] = *(const unsigned int*)&As[m1 + lr][klA];
      // B 32x16 bf16: same K order as A: VGPR 0-3 -> K 0..7 (+8*hf), VGPR 4-7 -> K 16..23 (+8*hf)
      int klB = ((j < 4) ? (2 * j) : (16 + 2 * (j - 4))) + 8 * hf;  // even
      bu.u[j] = *(const unsigned int*)&BsT[nBase + lr][klB];
    }
    c0 = __builtin_amdgcn_wmma_f32_16x16x32_bf16(false, a0.v, false, bu.v, (short)0, c0,
                                                 false, false);
    c1 = __builtin_amdgcn_wmma_f32_16x16x32_bf16(false, a1.v, false, bu.v, (short)0, c1,
                                                 false, false);
    __syncthreads();
  }

  // C/D 16x16 f32: VGPR r, lanes 0-15 -> M=r,N=lane; lanes 16-31 -> M=8+r
#pragma unroll
  for (int r = 0; r < 8; ++r) {
    int row0 = mBlock + m0 + r + 8 * hf;
    int row1 = mBlock + m1 + r + 8 * hf;
    float v0 = c0[r], v1 = c1[r];
    C[(long)row0 * N + nBase + lr] = v0 > 0.f ? v0 : 0.f;
    C[(long)row1 * N + nBase + lr] = v1 > 0.f ? v1 : 0.f;
  }
}

// tanh-attention softmax across 3 metapaths + L2 normalize + FC(128->8)
__global__ void fuse_k(const float* __restrict__ out_all, const float* __restrict__ attn,
                       const float* __restrict__ fc_w, const float* __restrict__ fc_b,
                       float* __restrict__ out, int B) {
  int b = blockIdx.x, d = threadIdx.x;   // 128 threads
  __shared__ float red[128];
  __shared__ float sc[4];
  __shared__ float semb[128];
  float v[3];
#pragma unroll
  for (int m = 0; m < 3; ++m) v[m] = out_all[((long)m * B + b) * D_FEAT + d];
#pragma unroll
  for (int m = 0; m < 3; ++m) {
    red[d] = v[m] * attn[d];
    __syncthreads();
    for (int off = 64; off > 0; off >>= 1) {
      if (d < off) red[d] += red[d + off];
      __syncthreads();
    }
    if (d == 0) sc[m] = tanhf(red[0]);
    __syncthreads();
  }
  float mx = fmaxf(sc[0], fmaxf(sc[1], sc[2]));
  float w0 = expf(sc[0] - mx), w1 = expf(sc[1] - mx), w2 = expf(sc[2] - mx);
  float ws = w0 + w1 + w2;
  float emb = (w0 * v[0] + w1 * v[1] + w2 * v[2]) / ws;
  red[d] = emb * emb;
  __syncthreads();
  for (int off = 64; off > 0; off >>= 1) {
    if (d < off) red[d] += red[d + off];
    __syncthreads();
  }
  if (d == 0) sc[3] = fmaxf(sqrtf(red[0]), 1e-12f);
  __syncthreads();
  semb[d] = emb / sc[3];
  __syncthreads();
  if (d < 8) {
    float acc = fc_b[d];
    for (int k = 0; k < D_FEAT; ++k) acc += semb[k] * fc_w[k * 8 + d];
    out[(long)b * 8 + d] = acc;
  }
}

extern "C" void kernel_launch(void* const* d_in, const int* in_sizes, int n_in,
                              void* d_out, int out_size, void* d_ws, size_t ws_size,
                              hipStream_t stream) {
  (void)n_in; (void)out_size; (void)ws_size;
  const int*   ids    = (const int*)  d_in[0];
  const float* feats  = (const float*)d_in[1];
  const float* eemb   = (const float*)d_in[2];
  const int*   adjN   = (const int*)  d_in[3];
  const int*   adjE   = (const int*)  d_in[4];
  const float* Wself  = (const float*)d_in[5];
  const float* Wneigh = (const float*)d_in[6];
  const float* Wedge  = (const float*)d_in[7];
  const float* attn   = (const float*)d_in[8];
  const float* fcw    = (const float*)d_in[9];
  const float* fcb    = (const float*)d_in[10];
  float* out = (float*)d_out;

  const int B = in_sizes[0];
  const int D = 128, E = 32, S = 16, NMP = 3, NL = 2;
  const int N_NODES = in_sizes[1] / D;
  const int N_EDGE  = in_sizes[2] / (NMP * E);
  const int N1 = B * S;       // 16384
  const int N2 = N1 * S;      // 262144 (index arrays only; level-2 feats never materialized)

  // workspace carve-up (~37 MB)
  char* wp = (char*)d_ws;
  auto alloc = [&](size_t bytes) -> char* {
    char* p = wp; wp += (bytes + 255) & ~(size_t)255; return p;
  };
  float* out_all = (float*)alloc((size_t)NMP * B * D * 4);
  int*   nb1   = (int*)  alloc((size_t)N1 * 4);
  int*   eid1  = (int*)  alloc((size_t)N1 * 4);
  int*   nb2   = (int*)  alloc((size_t)N2 * 4);
  int*   eid2  = (int*)  alloc((size_t)N2 * 4);
  float* feat0 = (float*)alloc((size_t)B * D * 4);
  float* feat1 = (float*)alloc((size_t)N1 * D * 4);
  float* edge1 = (float*)alloc((size_t)N1 * E * 4);
  float* nin0  = (float*)alloc((size_t)B * (D + E) * 4);
  float* nin1  = (float*)alloc((size_t)N1 * (D + E) * 4);
  float* h0    = (float*)alloc((size_t)B * D * 4);
  float* h1    = (float*)alloc((size_t)N1 * D * 4);
  float* e0p   = (float*)alloc((size_t)N1 * E * 4);
  float* nin0b = (float*)alloc((size_t)B * (D + E) * 4);

  gather_rows4_k<<<(B * 32 + 255) / 256, 256, 0, stream>>>(
      (const float4*)feats, ids, (float4*)feat0, B, 5);

  for (int mp = 0; mp < NMP; ++mp) {
    const int*   aN  = adjN + (size_t)mp * N_NODES * DEG_C;
    const int*   aE  = adjE + (size_t)mp * N_NODES * DEG_C;
    const float* em  = eemb + (size_t)mp * N_EDGE * E;
    const float* Ws0 = Wself  + ((size_t)mp * NL + 0) * D * D;
    const float* Ws1 = Wself  + ((size_t)mp * NL + 1) * D * D;
    const float* Wn0 = Wneigh + ((size_t)mp * NL + 0) * (D + E) * D;
    const float* Wn1 = Wneigh + ((size_t)mp * NL + 1) * (D + E) * D;
    const float* We0 = Wedge  + ((size_t)mp * NL + 0) * (2 * D + E) * E;

    sample_k<<<(B * S + 255) / 256, 256, 0, stream>>>(ids, B, aN, aE, nb1, eid1);
    gather_rows4_k<<<(N1 * 32 + 255) / 256, 256, 0, stream>>>(
        (const float4*)feats, nb1, (float4*)feat1, N1, 5);
    gather_rows4_k<<<(N1 * 8 + 255) / 256, 256, 0, stream>>>(
        (const float4*)em, eid1, (float4*)edge1, N1, 3);
    sample_k<<<(N1 * S + 255) / 256, 256, 0, stream>>>(nb1, N1, aN, aE, nb2, eid2);
    // level-2 gather fused into the mean (tables are L2-resident)
    mean_concat_k<<<N1, 64, 0, stream>>>(feats, nb2, em, eid2, nin1);
    mean_concat_k<<<B,  64, 0, stream>>>(feat1, nullptr, edge1, nullptr, nin0);

    // h0 = relu([feat0|nin0] @ [Ws0;Wn0])   M=B, K=288, N=128, Mtile=32
    gemm_relu_wmma_k<<<B / 32, 256, 0, stream>>>(feat0, D, 0, nin0, D + E, 0,
                                                 nullptr, 0, 0, Ws0, D, Wn0, h0, B, D);
    // h1 = relu([feat1|nin1] @ [Ws0;Wn0])   M=N1
    gemm_relu_wmma_k<<<N1 / 32, 256, 0, stream>>>(feat1, D, 0, nin1, D + E, 0,
                                                  nullptr, 0, 0, Ws0, D, Wn0, h1, N1, D);
    // e0' = relu([rep(h0,16)|h1|edge1] @ We0)  M=N1, N=32, K=288, Mtile=128
    gemm_relu_wmma_k<<<N1 / 128, 256, 0, stream>>>(h0, D, 4, h1, D, 0, edge1, E, 0,
                                                   We0, 2 * D + E, nullptr, e0p, N1, E);
    // layer 1
    mean_concat_k<<<B, 64, 0, stream>>>(h1, nullptr, e0p, nullptr, nin0b);
    gemm_relu_wmma_k<<<B / 32, 256, 0, stream>>>(h0, D, 0, nin0b, D + E, 0,
                                                 nullptr, 0, 0, Ws1, D, Wn1,
                                                 out_all + (size_t)mp * B * D, B, D);
  }

  fuse_k<<<B, 128, 0, stream>>>(out_all, attn, fcw, fcb, out, B);
}
